// QuanvolutionGen196_65481071395555
// MI455X (gfx1250) — compile-verified, matrix operand order checked by reading, not run on hardware
//
#include <hip/hip_runtime.h>
#include <hip/hip_bf16.h>

// ---------------------------------------------------------------------------
// MI455X (gfx1250) implementation.
// The network is a chain of huge GEMMs (~580 GFLOP) with tiny elementwise
// glue. Everything is cast once to f16 and run through
// v_wmma_f32_16x16x32_f16 (f32 accumulate), with RBF-exp / threshold / tanh
// epilogues fused into the GEMMs so intermediates stay f16 and chain directly
// into the next WMMA. Tile staging uses gfx1250 async global->LDS copies
// (ASYNCcnt) so waves keep issuing WMMAs while tiles land in LDS.
// ---------------------------------------------------------------------------

typedef _Float16 h8  __attribute__((ext_vector_type(8)));
typedef _Float16 h16 __attribute__((ext_vector_type(16)));
typedef float    f8  __attribute__((ext_vector_type(8)));

#define BDIM   8192
#define PDIM   8192
#define DRAW   784
#define DPAD   896          // 784 padded to multiple of 128 (zero filled)
#define H1DIM  1024
#define H2DIM  512
#define GAMMA  0.001f
#define THRESH 0.8f
#define LDSK   40           // padded LDS row stride (halves) for bank spread

__device__ __forceinline__ f8 wmma_f16(h16 a, h16 b, f8 c) {
  // emits v_wmma_f32_16x16x32_f16
  return __builtin_amdgcn_wmma_f32_16x16x32_f16(
      false, a, false, b, (short)0, c, false, false);
}

// Async global -> LDS 16-byte copy (gfx1250). LDS address is the low 32 bits
// of the generic pointer.
__device__ __forceinline__ void async_ld_b128(void* lds, const void* g) {
  uint32_t l = (uint32_t)(uintptr_t)lds;
  uint64_t a = (uint64_t)(uintptr_t)g;
  asm volatile("global_load_async_to_lds_b128 %0, %1, off"
               :: "v"(l), "v"(a) : "memory");
}
__device__ __forceinline__ void wait_async0() {
#if __has_builtin(__builtin_amdgcn_s_wait_asynccnt)
  __builtin_amdgcn_s_wait_asynccnt(0);
#else
  asm volatile("s_wait_asynccnt 0x0" ::: "memory");
#endif
}

// A/B fragment per ISA 16-bit layout: lane L holds row (L%16); its 16 halves
// are K-chunks [8h,8h+8) and [16+8h,16+8h+8) with h = L/16.
__device__ __forceinline__ h16 ldfrag(const _Float16* rowp, int h) {
  h8 lo = *(const h8*)(rowp + 8 * h);
  h8 hi = *(const h8*)(rowp + 16 + 8 * h);
  return __builtin_shufflevector(lo, hi, 0, 1, 2, 3, 4, 5, 6, 7,
                                 8, 9, 10, 11, 12, 13, 14, 15);
}

// ---------------------------------------------------------------------------
// Tiled WMMA GEMM:  C[M,N] = epi( A[M,K] * B )
//   B_KN = false: B stored [N,K] row-major (i.e. C = A * B^T)   (RBF GEMMs)
//   B_KN = true : B stored [K,N] row-major (normal GEMM)
// EPI: 0 = store f16
//      1 = exp(-g*(a2[m]+b2[n]-2*acc))                 (k_mat)
//      2 = (exp(...) >= THRESH) ? 1 : 0                (adj)
//      3 = tanh(acc + bias[n])                         (hidden layers)
// Tile: 128x128 per 256-thread block (8 wave32s; each wave 32x64 = 2x4 frags).
// ---------------------------------------------------------------------------
template <int EPI, bool B_KN>
__global__ __launch_bounds__(256) void gemm_wmma(
    const _Float16* __restrict__ A, const _Float16* __restrict__ B,
    _Float16* __restrict__ C,
    const float* __restrict__ rowsq, const float* __restrict__ colsq,
    const float* __restrict__ bias,
    int M, int N, int K, int lda, int ldb, int ldc) {
  __shared__ _Float16 As[2][128 * LDSK];
  __shared__ _Float16 Bs[2][128 * LDSK];

  const int m0   = blockIdx.y * 128;
  const int n0   = blockIdx.x * 128;
  const int tid  = threadIdx.x;
  const int lane = tid & 31;
  const int wave = tid >> 5;
  const int wm   = wave & 3;   // 4 waves along M (32 rows each)
  const int wn   = wave >> 2;  // 2 waves along N (64 cols each)
  const int lr   = lane & 15;
  const int lh   = lane >> 4;

  f8 acc[2][4];
#pragma unroll
  for (int i = 0; i < 2; ++i)
#pragma unroll
    for (int j = 0; j < 4; ++j) acc[i][j] = 0.0f;

  auto loadA = [&](int buf, int kb) {
#pragma unroll
    for (int t = 0; t < 2; ++t) {
      int c   = tid + t * 256;      // 512 chunks of 8 halves = 128x32 tile
      int row = c >> 2;
      int kc  = (c & 3) * 8;
      async_ld_b128(&As[buf][row * LDSK + kc],
                    A + (size_t)(m0 + row) * lda + kb + kc);
    }
  };
  auto loadB = [&](int buf, int kb) {
    if constexpr (!B_KN) {
#pragma unroll
      for (int t = 0; t < 2; ++t) {
        int c   = tid + t * 256;
        int row = c >> 2;
        int kc  = (c & 3) * 8;
        async_ld_b128(&Bs[buf][row * LDSK + kc],
                      B + (size_t)(n0 + row) * ldb + kb + kc);
      }
    } else {
#pragma unroll
      for (int t = 0; t < 2; ++t) {
        int c    = tid + t * 256;   // 32(K) x 128(N) tile, transpose into LDS
        int krow = c >> 4;
        int nc   = (c & 15) * 8;
        uint4 v  = *(const uint4*)(B + (size_t)(kb + krow) * ldb + n0 + nc);
        const _Float16* hv = (const _Float16*)&v;
#pragma unroll
        for (int j = 0; j < 8; ++j)
          Bs[buf][(nc + j) * LDSK + krow] = hv[j];
      }
    }
  };

  const int ksteps = K >> 5;
  loadA(0, 0);
  loadB(0, 0);
  wait_async0();
  __syncthreads();

  for (int kt = 0; kt < ksteps; ++kt) {
    const int cur = kt & 1;
    if (kt + 1 < ksteps) {  // async-prefetch next tile into the other buffer
      loadA(cur ^ 1, (kt + 1) << 5);
      loadB(cur ^ 1, (kt + 1) << 5);
    }
    h16 af[2];
#pragma unroll
    for (int fm = 0; fm < 2; ++fm)
      af[fm] = ldfrag(&As[cur][(wm * 32 + fm * 16 + lr) * LDSK], lh);
#pragma unroll
    for (int fn = 0; fn < 4; ++fn) {
      h16 bfr = ldfrag(&Bs[cur][(wn * 64 + fn * 16 + lr) * LDSK], lh);
      acc[0][fn] = wmma_f16(af[0], bfr, acc[0][fn]);
      acc[1][fn] = wmma_f16(af[1], bfr, acc[1][fn]);
    }
    wait_async0();   // own async copies into buf^1 done before barrier
    __syncthreads();
  }

  // Epilogue: C frag layout -> row = r + 8*(lane/16), col = lane%16
#pragma unroll
  for (int fm = 0; fm < 2; ++fm)
#pragma unroll
    for (int fn = 0; fn < 4; ++fn)
#pragma unroll
      for (int r = 0; r < 8; ++r) {
        int row = m0 + wm * 32 + fm * 16 + r + 8 * lh;
        int col = n0 + wn * 64 + fn * 16 + lr;
        float v = acc[fm][fn][r];
        if (EPI == 1 || EPI == 2) {
          float d2 = fmaxf(rowsq[row] + colsq[col] - 2.0f * v, 0.0f);
          v = __expf(-GAMMA * d2);
          if (EPI == 2) v = (v >= THRESH) ? 1.0f : 0.0f;
        } else if (EPI == 3) {
          v = tanhf(v + bias[col]);
        }
        C[(size_t)row * ldc + col] = (_Float16)v;
      }
}

// ---------------------------------------------------------------------------
// Conv2d(1,4,k=2,s=2) + flatten + f16 cast (padded to 896) + row sum-of-squares
// One block per image; memory bound.
// ---------------------------------------------------------------------------
__global__ __launch_bounds__(256) void conv_prep(
    const float* __restrict__ x, const float* __restrict__ w,
    const float* __restrict__ cb, _Float16* __restrict__ cf,
    float* __restrict__ a2) {
  __shared__ float red[256];
  const int b = blockIdx.x;
  const float* img = x + (size_t)b * 784;
  float ss = 0.0f;
  for (int o = threadIdx.x; o < DPAD; o += 256) {
    float v = 0.0f;
    if (o < DRAW) {
      int c = o / 196, rem = o % 196, y = rem / 14, xx = rem % 14;
      v = cb[c];
#pragma unroll
      for (int i = 0; i < 2; ++i)
#pragma unroll
        for (int j = 0; j < 2; ++j)
          v += w[c * 4 + i * 2 + j] * img[(2 * y + i) * 28 + (2 * xx + j)];
      ss += v * v;
    }
    cf[(size_t)b * DPAD + o] = (_Float16)v;
  }
  red[threadIdx.x] = ss;
  __syncthreads();
  for (int s = 128; s > 0; s >>= 1) {
    if (threadIdx.x < s) red[threadIdx.x] += red[threadIdx.x + s];
    __syncthreads();
  }
  if (threadIdx.x == 0) a2[b] = red[0];
}

// prototypes: f32[P,784] -> f16[P,896] (zero-padded) + row sum-of-squares
__global__ __launch_bounds__(256) void row_prep(
    const float* __restrict__ src, _Float16* __restrict__ dst,
    float* __restrict__ sq) {
  __shared__ float red[256];
  const int p = blockIdx.x;
  float ss = 0.0f;
  for (int o = threadIdx.x; o < DPAD; o += 256) {
    float v = 0.0f;
    if (o < DRAW) {
      v = src[(size_t)p * DRAW + o];
      ss += v * v;
    }
    dst[(size_t)p * DPAD + o] = (_Float16)v;
  }
  red[threadIdx.x] = ss;
  __syncthreads();
  for (int s = 128; s > 0; s >>= 1) {
    if (threadIdx.x < s) red[threadIdx.x] += red[threadIdx.x + s];
    __syncthreads();
  }
  if (threadIdx.x == 0) sq[p] = red[0];
}

// generic f32 -> f16 cast with zero row padding (for W1, W2)
__global__ __launch_bounds__(256) void pad_cast(
    const float* __restrict__ src, _Float16* __restrict__ dst,
    int rows_valid, int rows_total, int cols) {
  size_t i = (size_t)blockIdx.x * 256 + threadIdx.x;
  size_t tot = (size_t)rows_total * cols;
  if (i >= tot) return;
  int r = (int)(i / cols);
  int c = (int)(i % cols);
  dst[i] = (r < rows_valid) ? (_Float16)src[(size_t)r * cols + c]
                            : (_Float16)0.0f;
}

// head: logits = h2 @ Wf + bf ; log_softmax. One thread per batch row (tiny).
__global__ __launch_bounds__(256) void head_kernel(
    const _Float16* __restrict__ h2, const float* __restrict__ Wf,
    const float* __restrict__ bf, float* __restrict__ out) {
  int b = blockIdx.x * 256 + threadIdx.x;
  if (b >= BDIM) return;
  float l[10];
#pragma unroll
  for (int j = 0; j < 10; ++j) l[j] = bf[j];
  for (int k = 0; k < H2DIM; ++k) {
    float hv = (float)h2[(size_t)b * H2DIM + k];
#pragma unroll
    for (int j = 0; j < 10; ++j) l[j] += hv * Wf[k * 10 + j];
  }
  float m = l[0];
#pragma unroll
  for (int j = 1; j < 10; ++j) m = fmaxf(m, l[j]);
  float s = 0.0f;
#pragma unroll
  for (int j = 0; j < 10; ++j) s += __expf(l[j] - m);
  float lse = m + __logf(s);
#pragma unroll
  for (int j = 0; j < 10; ++j) out[(size_t)b * 10 + j] = l[j] - lse;
}

// ---------------------------------------------------------------------------
extern "C" void kernel_launch(void* const* d_in, const int* in_sizes, int n_in,
                              void* d_out, int out_size, void* d_ws,
                              size_t ws_size, hipStream_t stream) {
  const float* x      = (const float*)d_in[0];
  const float* conv_w = (const float*)d_in[1];
  const float* conv_b = (const float*)d_in[2];
  const float* protos = (const float*)d_in[3];
  const float* W1     = (const float*)d_in[4];
  const float* b1     = (const float*)d_in[5];
  const float* W2     = (const float*)d_in[6];
  const float* b2     = (const float*)d_in[7];
  const float* Wf     = (const float*)d_in[8];
  const float* bf     = (const float*)d_in[9];
  float* out = (float*)d_out;

  char* ws = (char*)d_ws;
  size_t off = 0;
  auto carve = [&](size_t bytes) -> void* {
    void* p = ws + off;
    off += (bytes + 255) & ~(size_t)255;
    return p;
  };

  _Float16* cf16  = (_Float16*)carve((size_t)BDIM * DPAD * 2);
  float*    a2    = (float*)carve((size_t)BDIM * 4);
  _Float16* pf16  = (_Float16*)carve((size_t)PDIM * DPAD * 2);
  float*    p2    = (float*)carve((size_t)PDIM * 4);
  _Float16* w1f   = (_Float16*)carve((size_t)DPAD * H1DIM * 2);
  _Float16* w2f   = (_Float16*)carve((size_t)H1DIM * H2DIM * 2);
  _Float16* kmatf = (_Float16*)carve((size_t)BDIM * PDIM * 2);
  _Float16* adjf  = (_Float16*)carve((size_t)PDIM * PDIM * 2);
  _Float16* featf = (_Float16*)carve((size_t)BDIM * DPAD * 2);
  _Float16* gf    = (_Float16*)carve((size_t)BDIM * DPAD * 2);
  _Float16* h1f   = (_Float16*)carve((size_t)BDIM * H1DIM * 2);
  _Float16* t2f   = (_Float16*)carve((size_t)BDIM * H1DIM * 2);
  _Float16* h2f   = (_Float16*)carve((size_t)BDIM * H2DIM * 2);
  (void)ws_size; (void)in_sizes; (void)n_in; (void)out_size;

  // ---- prep: conv + casts + squared norms --------------------------------
  conv_prep<<<BDIM, 256, 0, stream>>>(x, conv_w, conv_b, cf16, a2);
  row_prep<<<PDIM, 256, 0, stream>>>(protos, pf16, p2);
  {
    size_t t1 = (size_t)DPAD * H1DIM;
    pad_cast<<<(int)((t1 + 255) / 256), 256, 0, stream>>>(W1, w1f, DRAW, DPAD,
                                                          H1DIM);
    size_t t2 = (size_t)H1DIM * H2DIM;
    pad_cast<<<(int)((t2 + 255) / 256), 256, 0, stream>>>(W2, w2f, H1DIM,
                                                          H1DIM, H2DIM);
  }

  // ---- k_mat = exp(-g*(|a|^2 + |p|^2 - 2 A P^T))  [8192x8192] ------------
  gemm_wmma<1, false><<<dim3(PDIM / 128, BDIM / 128), 256, 0, stream>>>(
      cf16, pf16, kmatf, a2, p2, nullptr, BDIM, PDIM, DPAD, DPAD, DPAD, PDIM);

  // ---- adj = (exp(-g*d2(P,P)) >= 0.8)  [8192x8192] -----------------------
  gemm_wmma<2, false><<<dim3(PDIM / 128, PDIM / 128), 256, 0, stream>>>(
      pf16, pf16, adjf, p2, p2, nullptr, PDIM, PDIM, DPAD, DPAD, DPAD, PDIM);

  // ---- features = k_mat @ protos  [8192x896] -----------------------------
  gemm_wmma<0, true><<<dim3(DPAD / 128, BDIM / 128), 256, 0, stream>>>(
      kmatf, pf16, featf, nullptr, nullptr, nullptr, BDIM, DPAD, PDIM, PDIM,
      DPAD, DPAD);

  // ---- g = adj @ features  [8192x896] ------------------------------------
  gemm_wmma<0, true><<<dim3(DPAD / 128, BDIM / 128), 256, 0, stream>>>(
      adjf, featf, gf, nullptr, nullptr, nullptr, BDIM, DPAD, PDIM, PDIM, DPAD,
      DPAD);

  // ---- h1 = tanh(g @ W1 + b1)  [8192x1024] -------------------------------
  gemm_wmma<3, true><<<dim3(H1DIM / 128, BDIM / 128), 256, 0, stream>>>(
      gf, w1f, h1f, nullptr, nullptr, b1, BDIM, H1DIM, DPAD, DPAD, H1DIM,
      H1DIM);

  // ---- t2 = adj @ h1  [8192x1024] ----------------------------------------
  gemm_wmma<0, true><<<dim3(H1DIM / 128, BDIM / 128), 256, 0, stream>>>(
      adjf, h1f, t2f, nullptr, nullptr, nullptr, BDIM, H1DIM, PDIM, PDIM,
      H1DIM, H1DIM);

  // ---- h2 = tanh(t2 @ W2 + b2)  [8192x512] -------------------------------
  gemm_wmma<3, true><<<dim3(H2DIM / 128, BDIM / 128), 256, 0, stream>>>(
      t2f, w2f, h2f, nullptr, nullptr, b2, BDIM, H2DIM, H1DIM, H1DIM, H2DIM,
      H2DIM);

  // ---- logits + log_softmax ----------------------------------------------
  head_kernel<<<(BDIM + 255) / 256, 256, 0, stream>>>(h2f, Wf, bf, out);
}
